// SimpleLanePredictor_65927747994072
// MI455X (gfx1250) — matrix operand order repaired and checked by reading, not compiled, not run on hardware
//
#include <hip/hip_runtime.h>

#define N_LANES 16
#define HID     64
#define OUTD    256
#define ROW_ITERS 4     // 8 waves * 16 rows * 4 iters = 512 rows per block
#define WGT_HALVES (4*2*32*16 + 16*2*32*16)   // 4096 + 16384 = 20480 per lane-net

typedef __attribute__((ext_vector_type(16))) _Float16 v16h;
typedef __attribute__((ext_vector_type(8)))  _Float16 v8h;
typedef __attribute__((ext_vector_type(8)))  float    v8f;
typedef __attribute__((ext_vector_type(4)))  unsigned int v4u;

// Load a 16-half WMMA operand fragment from LDS (two ds_load_b128).
__device__ inline v16h load_frag16(const _Float16* p) {
    v8h lo = *(const v8h*)p;
    v8h hi = *(const v8h*)(p + 8);
    v16h r;
#pragma unroll
    for (int i = 0; i < 8; ++i) { r[i] = lo[i]; r[i + 8] = hi[i]; }
    return r;
}

__device__ inline v8f wmma_f16(v16h a, v16h b, v8f c) {
    return __builtin_amdgcn_wmma_f32_16x16x32_f16(
        false, a, false, b, (short)0, c, false, false);
}

// Swizzle decode: linear index e over [frag_id][lane][j], frag_id = nt*2 + t.
// Element j of lane ls is (k = t*32 + (j>>3)*16 + (ls>>4)*8 + (j&7), n = nt*16 + (ls&15)).
__device__ inline void decode_frag(int e, int& k, int& n) {
    int j  = e & 15;
    int ls = (e >> 4) & 31;
    int t  = (e >> 9) & 1;
    int nt = e >> 10;
    k = t * 32 + ((j >> 3) << 4) + ((ls >> 4) << 3) + (j & 7);
    n = nt * 16 + (ls & 15);
}

// One-time per launch: build swizzled f16 weight images in workspace.
__global__ __launch_bounds__(256)
void prep_weights(const float* __restrict__ W2, const float* __restrict__ W3,
                  _Float16* __restrict__ ws)
{
    const int l   = blockIdx.x;
    const int tid = threadIdx.x;
    const float* gW2 = W2 + (size_t)l * HID * HID;
    const float* gW3 = W3 + (size_t)l * HID * OUTD;
    _Float16* w = ws + (size_t)l * WGT_HALVES;
    for (int e = tid; e < 4096; e += 256) {
        int k, n; decode_frag(e, k, n);
        w[e] = (_Float16)gW2[k * HID + n];
    }
    for (int e = tid; e < 16384; e += 256) {
        int k, n; decode_frag(e, k, n);
        w[4096 + e] = (_Float16)gW3[k * OUTD + n];
    }
}

__global__ __launch_bounds__(256)
void lane_mlp_kernel(const float* __restrict__ x,
                     const float* __restrict__ W1, const float* __restrict__ b1,
                     const float* __restrict__ W2, const float* __restrict__ b2,
                     const float* __restrict__ W3, const float* __restrict__ b3,
                     const _Float16* __restrict__ wsw,   // swizzled weights or null
                     float* __restrict__ out, int B)
{
    __shared__ __align__(16) _Float16 sWgt[WGT_HALVES];  // sW2 | sW3, 40 KB
    __shared__ _Float16 sH2[8][16 * 64];                 // 16 KB per-wave h2 staging
    __shared__ float sw1[HID], sb1[HID], sb2[HID];
    __shared__ float sb3[OUTD];

    const int tid  = threadIdx.x;
    const int lane = tid & 31;
    const int wave = tid >> 5;
    const int hi   = lane >> 4;
    const int r16  = lane & 15;
    const int l    = blockIdx.y;

    _Float16* const sW2 = sWgt;
    _Float16* const sW3 = sWgt + 4096;

    // ---- weight staging ----
    if (wsw) {
        const v4u* src = (const v4u*)(wsw + (size_t)l * WGT_HALVES);
        v4u* dst = (v4u*)sWgt;
        for (int e = tid; e < WGT_HALVES / 8; e += 256) dst[e] = src[e];
    } else {
        const float* gW2 = W2 + (size_t)l * HID * HID;
        const float* gW3 = W3 + (size_t)l * HID * OUTD;
        for (int e = tid; e < 4096; e += 256) {
            int k, n; decode_frag(e, k, n);
            sW2[e] = (_Float16)gW2[k * HID + n];
        }
        for (int e = tid; e < 16384; e += 256) {
            int k, n; decode_frag(e, k, n);
            sW3[e] = (_Float16)gW3[k * OUTD + n];
        }
    }
    if (tid < HID) {
        sw1[tid] = W1[l * HID + tid];
        sb1[tid] = b1[l * HID + tid];
        sb2[tid] = b2[l * HID + tid];
    }
    sb3[tid] = b3[l * OUTD + tid];
    __syncthreads();

    // ---- hoist only cheap scalars (keep VGPRs well under 256) ----
    v16h w1f[2], b1f[2];   // W1/b1 pre-swizzled to A-fragment layout: 32 VGPRs
#pragma unroll
    for (int t = 0; t < 2; ++t) {
#pragma unroll
        for (int j = 0; j < 16; ++j) {
            int k = t * 32 + ((j >> 3) << 4) + (hi << 3) + (j & 7);
            w1f[t][j] = (_Float16)sw1[k];
            b1f[t][j] = (_Float16)sb1[k];
        }
    }
    float bias2[4];
#pragma unroll
    for (int nt = 0; nt < 4; ++nt) bias2[nt] = sb2[nt * 16 + r16];
    float bias3[16];
#pragma unroll
    for (int nt = 0; nt < 16; ++nt) bias3[nt] = sb3[nt * 16 + r16];

    const int block_row0 = blockIdx.x * (128 * ROW_ITERS);

    for (int it = 0; it < ROW_ITERS; ++it) {
        const int row0 = block_row0 + it * 128 + wave * 16;

        // ---- layer 1: packed f16 math directly into A-fragment layout ----
        const float xv = __builtin_nontemporal_load(&x[(size_t)(row0 + r16) * N_LANES + l]);
        const _Float16 xh = (_Float16)xv;
        v16h xs;
#pragma unroll
        for (int j = 0; j < 16; ++j) xs[j] = xh;
        v16h a1[2];
#pragma unroll
        for (int t = 0; t < 2; ++t) {
            v16h h = xs * w1f[t] + b1f[t];                  // v_pk_fma_f16
            a1[t] = __builtin_elementwise_max(h, (v16h)0);  // v_pk_max relu
        }

        // ---- layer 2: 4 accumulators, C starts as inline 0 (no init regs) ----
        v8f c2[4] = {};
#pragma unroll
        for (int half = 0; half < 2; ++half) {
            const int t0 = half * 2, t1 = t0 + 1;
            v16h f00 = load_frag16(&sW2[(((t0 << 1) + 0) * 32 + lane) << 4]);
            v16h f01 = load_frag16(&sW2[(((t0 << 1) + 1) * 32 + lane) << 4]);
            v16h f10 = load_frag16(&sW2[(((t1 << 1) + 0) * 32 + lane) << 4]);
            v16h f11 = load_frag16(&sW2[(((t1 << 1) + 1) * 32 + lane) << 4]);
            c2[t0] = wmma_f16(a1[0], f00, c2[t0]);
            c2[t0] = wmma_f16(a1[1], f01, c2[t0]);
            c2[t1] = wmma_f16(a1[0], f10, c2[t1]);
            c2[t1] = wmma_f16(a1[1], f11, c2[t1]);
        }
        // bias + relu + f16, stage to per-wave LDS tile (row-major [m][k]).
#pragma unroll
        for (int nt = 0; nt < 4; ++nt) {
#pragma unroll
            for (int i = 0; i < 8; ++i) {
                float v = fmaxf(c2[nt][i] + bias2[nt], 0.0f);
                sH2[wave][(i + (hi << 3)) * 64 + nt * 16 + r16] = (_Float16)v;
            }
        }
        // Within-wave DS ops are in-order; reload as A fragments (2x b128 each).
        v16h a2[2];
#pragma unroll
        for (int t = 0; t < 2; ++t) {
            const _Float16* hp = &sH2[wave][r16 * 64 + t * 32 + (hi << 3)];
            v8h lo = *(const v8h*)hp;
            v8h hh = *(const v8h*)(hp + 16);
#pragma unroll
            for (int i = 0; i < 8; ++i) { a2[t][i] = lo[i]; a2[t][i + 8] = hh[i]; }
        }

        // ---- layer 3: two N-tiles in flight to hide WMMA->VALU hazard ----
        float* const obase = out + (size_t)row0 * (N_LANES * OUTD) + l * OUTD;
#pragma unroll
        for (int np = 0; np < 8; ++np) {
            const int nt0 = np * 2, nt1 = nt0 + 1;
            v16h b00 = load_frag16(&sW3[(((nt0 << 1) + 0) * 32 + lane) << 4]);
            v16h b01 = load_frag16(&sW3[(((nt0 << 1) + 1) * 32 + lane) << 4]);
            v16h b10 = load_frag16(&sW3[(((nt1 << 1) + 0) * 32 + lane) << 4]);
            v16h b11 = load_frag16(&sW3[(((nt1 << 1) + 1) * 32 + lane) << 4]);
            v8f c0 = {}, c1 = {};
            c0 = wmma_f16(a2[0], b00, c0);
            c0 = wmma_f16(a2[1], b01, c0);
            c1 = wmma_f16(a2[0], b10, c1);
            c1 = wmma_f16(a2[1], b11, c1);
#pragma unroll
            for (int i = 0; i < 8; ++i) {
                int m = i + (hi << 3);
                __builtin_nontemporal_store(c0[i] + bias3[nt0],
                    obase + (size_t)m * (N_LANES * OUTD) + nt0 * 16 + r16);
                __builtin_nontemporal_store(c1[i] + bias3[nt1],
                    obase + (size_t)m * (N_LANES * OUTD) + nt1 * 16 + r16);
            }
        }
    }
}

extern "C" void kernel_launch(void* const* d_in, const int* in_sizes, int n_in,
                              void* d_out, int out_size, void* d_ws, size_t ws_size,
                              hipStream_t stream) {
    const float* x  = (const float*)d_in[0];
    const float* W1 = (const float*)d_in[1];
    const float* b1 = (const float*)d_in[2];
    const float* W2 = (const float*)d_in[3];
    const float* b2 = (const float*)d_in[4];
    const float* W3 = (const float*)d_in[5];
    const float* b3 = (const float*)d_in[6];
    float* out = (float*)d_out;

    const int B = in_sizes[0] / N_LANES;             // 65536
    const int rows_per_block = 128 * ROW_ITERS;      // 512
    dim3 grid(B / rows_per_block, N_LANES);          // (128, 16)

    const size_t ws_need = (size_t)N_LANES * WGT_HALVES * sizeof(_Float16); // 640 KB
    if (ws_size >= ws_need) {
        _Float16* wsw = (_Float16*)d_ws;
        prep_weights<<<dim3(N_LANES), 256, 0, stream>>>(W2, W3, wsw);
        lane_mlp_kernel<<<grid, 256, 0, stream>>>(x, W1, b1, W2, b2, W3, b3, wsw, out, B);
    } else {
        lane_mlp_kernel<<<grid, 256, 0, stream>>>(x, W1, b1, W2, b2, W3, b3, nullptr, out, B);
    }
}
